// HinSAGE_14774687498450
// MI455X (gfx1250) — matrix operand order repaired and checked by reading, not compiled, not run on hardware
//
#include <hip/hip_runtime.h>
#include <hip/hip_bf16.h>

#define N_TX       200000
#define N_CLIENT   100000
#define N_MERCHANT 20000
#define NEDGE      1000000
#define DD         64

typedef __attribute__((ext_vector_type(16))) _Float16 v16h;
typedef __attribute__((ext_vector_type(8)))  float    v8f;

// ---------------------------------------------------------------------------
// edge gather + scatter-add: sum[dst] += H[src], fused degree count.
// 16 threads per edge, each thread moves one float4 (coalesced 256B row read
// per 16-lane group).  Lane c4==0 also bumps the degree counter (layer 0 only;
// degrees are layer-invariant).
// ---------------------------------------------------------------------------
__global__ void scatter_add(const float* __restrict__ H,
                            const int* __restrict__ src,
                            const int* __restrict__ dst,
                            int E, float* __restrict__ sum,
                            float* __restrict__ deg) {
    long long tid = (long long)blockIdx.x * blockDim.x + threadIdx.x;
    long long total = (long long)E * 16;
    if (tid >= total) return;
    int e  = (int)(tid >> 4);
    int c4 = (int)(tid & 15);
    int s = src[e], d = dst[e];
    const float4 v = *(const float4*)(H + (long long)s * DD + c4 * 4);
    float* o = sum + (long long)d * DD + c4 * 4;
    atomicAdd(o + 0, v.x);
    atomicAdd(o + 1, v.y);
    atomicAdd(o + 2, v.z);
    atomicAdd(o + 3, v.w);
    if (deg != nullptr && c4 == 0) atomicAdd(deg + d, 1.0f);
}

// ---------------------------------------------------------------------------
// A fragment (16-bit A 16x32, wave32; contiguous K order chosen here, B matches):
// lanes 0-15 row M=lane,   K in {kh*16+0..15} with kh=0
// lanes 16-31 same rows,   kh=1.
// One contiguous 64B run per lane -> 4 x global_load_b128.
// ---------------------------------------------------------------------------
__device__ __forceinline__ v16h load_a_frag(const float* __restrict__ H,
                                            int row, int kbase, float scale) {
    const int kh = (threadIdx.x & 31) >> 4;
    const float* p = H + (long long)row * DD + kbase + kh * 16;
    const float4 x0 = *(const float4*)(p + 0);
    const float4 x1 = *(const float4*)(p + 4);
    const float4 x2 = *(const float4*)(p + 8);
    const float4 x3 = *(const float4*)(p + 12);
    v16h a;
    a[0]  = (_Float16)(x0.x * scale); a[1]  = (_Float16)(x0.y * scale);
    a[2]  = (_Float16)(x0.z * scale); a[3]  = (_Float16)(x0.w * scale);
    a[4]  = (_Float16)(x1.x * scale); a[5]  = (_Float16)(x1.y * scale);
    a[6]  = (_Float16)(x1.z * scale); a[7]  = (_Float16)(x1.w * scale);
    a[8]  = (_Float16)(x2.x * scale); a[9]  = (_Float16)(x2.y * scale);
    a[10] = (_Float16)(x2.z * scale); a[11] = (_Float16)(x2.w * scale);
    a[12] = (_Float16)(x3.x * scale); a[13] = (_Float16)(x3.y * scale);
    a[14] = (_Float16)(x3.z * scale); a[15] = (_Float16)(x3.w * scale);
    return a;
}

// ---------------------------------------------------------------------------
// Fused SAGE GEMM: Out[i,:] = act( sum_t A_t[i,:]*scale_t @ W_t + bias0+bias1 )
// scale_t = 1/max(deg_t[i],1) when deg_t != null (mean aggregation), else 1.
//
// Block = 128 threads = 4 wave32s; wave w computes rows [blk*64+w*16, +16) x 64.
// Weights (up to 4 x 64x64 f32) are staged ONCE per block into LDS as f16,
// TRANSPOSED (WT[n][k]) so a B fragment (lane&15 = column n, lane>>4 selects
// K half) is 16 consecutive halves -> two ds_load_b128 per fragment.
// All row counts are multiples of 16, so the post-barrier early-exit is
// wave-uniform and EXEC is all-1s around every v_wmma (ISA 7.12 requirement).
// ---------------------------------------------------------------------------
__global__ __launch_bounds__(128) void sage_gemm(
    int N,
    const float* A0, const float* A1, const float* A2, const float* A3,
    const float* D0, const float* D1, const float* D2, const float* D3,
    const float* W0, const float* W1, const float* W2, const float* W3,
    int nterms,
    const float* bias0, const float* bias1,
    float* __restrict__ Out, int leaky) {

    __shared__ __attribute__((aligned(32))) _Float16 WT[4][DD * DD];

    const int tid  = threadIdx.x;
    const int wave = tid >> 5;
    const int lane = tid & 31;

    const float* As[4] = {A0, A1, A2, A3};
    const float* Ds[4] = {D0, D1, D2, D3};
    const float* Ws[4] = {W0, W1, W2, W3};

    // ---- stage weights to LDS: f32 -> f16, transposed ----
#pragma unroll
    for (int t = 0; t < 4; ++t) {
        if (t >= nterms) break;
        const float4* W4 = (const float4*)Ws[t];
        for (int i = tid; i < DD * DD / 4; i += 128) {     // 8 iters, b128 reads
            const float4 v = W4[i];
            const int base = i * 4;
            const int k = base >> 6;       // source row  (K index)
            const int n = base & 63;       // source cols (N index) n..n+3
            WT[t][(n + 0) * DD + k] = (_Float16)v.x;
            WT[t][(n + 1) * DD + k] = (_Float16)v.y;
            WT[t][(n + 2) * DD + k] = (_Float16)v.z;
            WT[t][(n + 3) * DD + k] = (_Float16)v.w;
        }
    }
    __syncthreads();

    const int rowBase = blockIdx.x * 64 + wave * 16;
    if (rowBase >= N) return;                     // wave-uniform, after barrier
    const int arow = rowBase + (lane & 15);
    const int bn   = lane & 15;                   // B fragment column within tile
    const int bks  = lane >> 4;                   // B fragment K-half (16 K values)

    v8f acc[4] = {};                              // 16x64 f32 accumulators

#pragma unroll
    for (int t = 0; t < 4; ++t) {
        if (t >= nterms) break;
        float scale = 1.0f;
        if (Ds[t]) scale = 1.0f / fmaxf(Ds[t][arow], 1.0f);
#pragma unroll
        for (int kt = 0; kt < 2; ++kt) {          // K = 64 -> two 32-wide tiles
            const v16h a = load_a_frag(As[t], arow, kt * 32, scale);
#pragma unroll
            for (int nt = 0; nt < 4; ++nt) {
                // B fragment: 16 consecutive halves from transposed LDS image
                const v16h b = *(const v16h*)&WT[t][(nt * 16 + bn) * DD + kt * 32 + bks * 16];
                acc[nt] = __builtin_amdgcn_wmma_f32_16x16x32_f16(
                    false, a, false, b, (short)0, acc[nt], false, false);
            }
        }
    }

    // Epilogue: C/D layout -> VGPR r: lanes0-15 row M=r, lanes16-31 row M=8+r
    const int khalf = lane >> 4;
#pragma unroll
    for (int nt = 0; nt < 4; ++nt) {
        const int n = nt * 16 + (lane & 15);
        float bv = 0.0f;
        if (bias0) bv += bias0[n];
        if (bias1) bv += bias1[n];
#pragma unroll
        for (int r = 0; r < 8; ++r) {
            float v = acc[nt][r] + bv;
            if (leaky) v = (v > 0.0f) ? v : 0.01f * v;
            const int row = rowBase + r + 8 * khalf;
            Out[(long long)row * DD + n] = v;
        }
    }
}

// ---------------------------------------------------------------------------
static inline void launch_scatter(const float* H, const int* src, const int* dst,
                                  float* sum, float* deg, hipStream_t s) {
    long long total = (long long)NEDGE * 16;
    int blocks = (int)((total + 255) / 256);
    scatter_add<<<blocks, 256, 0, s>>>(H, src, dst, NEDGE, sum, deg);
}

extern "C" void kernel_launch(void* const* d_in, const int* in_sizes, int n_in,
                              void* d_out, int out_size, void* d_ws, size_t ws_size,
                              hipStream_t stream) {
    const float* features = (const float*)d_in[0];
    const float* emb_client = (const float*)d_in[1];
    const float* emb_merchant = (const float*)d_in[2];
    const float* Ws_l0 = (const float*)d_in[3];   // [4,64,64]
    const float* Wn_l0 = (const float*)d_in[4];
    const float* b_l0  = (const float*)d_in[5];   // [4,64]
    const float* Ws_l1 = (const float*)d_in[6];
    const float* Wn_l1 = (const float*)d_in[7];
    const float* b_l1  = (const float*)d_in[8];
    const float* W_out = (const float*)d_in[9];
    const float* b_out = (const float*)d_in[10];
    const int* e0_src = (const int*)d_in[11];
    const int* e0_dst = (const int*)d_in[12];
    const int* e1_src = (const int*)d_in[13];
    const int* e1_dst = (const int*)d_in[14];
    const int* e2_src = (const int*)d_in[15];
    const int* e2_dst = (const int*)d_in[16];
    const int* e3_src = (const int*)d_in[17];
    const int* e3_dst = (const int*)d_in[18];

    float* out_proj = (float*)d_out;                       // [N_TX,64]
    float* emb      = (float*)d_out + (size_t)N_TX * DD;   // [N_TX,64]

    // ---- workspace partition (floats) ----
    float* ws = (float*)d_ws;
    size_t o = 0;
    float* sum0 = ws + o; o += (size_t)N_TX * DD;
    float* sum2 = ws + o; o += (size_t)N_TX * DD;
    float* sum1 = ws + o; o += (size_t)N_CLIENT * DD;
    float* sum3 = ws + o; o += (size_t)N_MERCHANT * DD;
    float* h1_tx = ws + o; o += (size_t)N_TX * DD;
    float* h1_cl = ws + o; o += (size_t)N_CLIENT * DD;
    float* h1_me = ws + o; o += (size_t)N_MERCHANT * DD;
    float* deg0 = ws + o; o += N_TX;
    float* deg2 = ws + o; o += N_TX;
    float* deg1 = ws + o; o += N_CLIENT;
    float* deg3 = ws + o; o += N_MERCHANT;

    const int W4096 = DD * DD;

    // ================= Layer 0 (degrees fused into scatter) =================
    hipMemsetAsync(deg0, 0, sizeof(float) * (2 * N_TX + N_CLIENT + N_MERCHANT), stream);
    hipMemsetAsync(sum0, 0, sizeof(float) * (size_t)DD * (2 * N_TX + N_CLIENT + N_MERCHANT), stream);
    launch_scatter(emb_client,   e0_src, e0_dst, sum0, deg0, stream);  // client -> tx
    launch_scatter(emb_merchant, e2_src, e2_dst, sum2, deg2, stream);  // merchant -> tx
    launch_scatter(features,     e1_src, e1_dst, sum1, deg1, stream);  // tx -> client
    launch_scatter(features,     e3_src, e3_dst, sum3, deg3, stream);  // tx -> merchant

    // tx: feats@Ws0 + feats@Ws2 + mean0@Wn0 + mean2@Wn2 + b0 + b2, leaky
    sage_gemm<<<(N_TX + 63) / 64, 128, 0, stream>>>(
        N_TX,
        features, features, sum0, sum2,
        nullptr, nullptr, deg0, deg2,
        Ws_l0 + 0 * W4096, Ws_l0 + 2 * W4096, Wn_l0 + 0 * W4096, Wn_l0 + 2 * W4096,
        4, b_l0 + 0 * DD, b_l0 + 2 * DD, h1_tx, 1);
    // client: emb_c@Ws1 + mean1@Wn1 + b1, leaky
    sage_gemm<<<(N_CLIENT + 63) / 64, 128, 0, stream>>>(
        N_CLIENT,
        emb_client, sum1, nullptr, nullptr,
        nullptr, deg1, nullptr, nullptr,
        Ws_l0 + 1 * W4096, Wn_l0 + 1 * W4096, nullptr, nullptr,
        2, b_l0 + 1 * DD, nullptr, h1_cl, 1);
    // merchant: emb_m@Ws3 + mean3@Wn3 + b3, leaky
    sage_gemm<<<(N_MERCHANT + 63) / 64, 128, 0, stream>>>(
        N_MERCHANT,
        emb_merchant, sum3, nullptr, nullptr,
        nullptr, deg3, nullptr, nullptr,
        Ws_l0 + 3 * W4096, Wn_l0 + 3 * W4096, nullptr, nullptr,
        2, b_l0 + 3 * DD, nullptr, h1_me, 1);

    // ================= Layer 1 (only tx output is live) =================
    hipMemsetAsync(sum0, 0, sizeof(float) * (size_t)DD * 2 * N_TX, stream);
    launch_scatter(h1_cl, e0_src, e0_dst, sum0, nullptr, stream);   // client -> tx
    launch_scatter(h1_me, e2_src, e2_dst, sum2, nullptr, stream);   // merchant -> tx

    // embedding = h1_tx@Ws1_0 + h1_tx@Ws1_2 + mean0@Wn1_0 + mean2@Wn1_2 + b (no act)
    sage_gemm<<<(N_TX + 63) / 64, 128, 0, stream>>>(
        N_TX,
        h1_tx, h1_tx, sum0, sum2,
        nullptr, nullptr, deg0, deg2,
        Ws_l1 + 0 * W4096, Ws_l1 + 2 * W4096, Wn_l1 + 0 * W4096, Wn_l1 + 2 * W4096,
        4, b_l1 + 0 * DD, b_l1 + 2 * DD, emb, 0);

    // out = emb @ W_out + b_out
    sage_gemm<<<(N_TX + 63) / 64, 128, 0, stream>>>(
        N_TX,
        emb, nullptr, nullptr, nullptr,
        nullptr, nullptr, nullptr, nullptr,
        W_out, nullptr, nullptr, nullptr,
        1, b_out, nullptr, out_proj, 0);
}
